// MLPAttention_154618823101
// MI455X (gfx1250) — matrix operand order repaired and checked by reading, not compiled, not run on hardware
//
#include <hip/hip_runtime.h>

namespace {

constexpr int Bn = 4;
constexpr int Ln = 4096;
constexpr int DIMn = 1024;
constexpr int Hn = 16;
constexpr int Dh = 64;        // DQK = DV = DINNER
constexpr int CH = 64;        // chunk length
constexpr int NC = Ln / CH;   // 64 chunks
constexpr float LR0 = 0.001f;
constexpr float WD0 = 0.9f;

typedef __attribute__((ext_vector_type(16))) __bf16 v16bf;
typedef __attribute__((ext_vector_type(8)))  float  v8f;

union FragU {
  v16bf v;
  unsigned short u[16];
  uint4 q[2];
};

__device__ __forceinline__ unsigned short f2bf(float f) {
  const __bf16 h = (__bf16)f;                 // native f32->bf16 convert (RNE)
  return __builtin_bit_cast(unsigned short, h);
}

// ---- WMMA fragment loaders (CDNA5 ISA 7.12.2 layouts, bf16 16x16x32) ----
// A (16x32, MxK): lane l -> M=l%16, half=l/16; vgpr j holds K pair
//   k = 16*(j/4) + 8*half + 2*(j%4)
// => per-lane data is two contiguous 8-element runs: [k0+8h .. +7] and [+16 ..]
__device__ __forceinline__ v16bf ldA(const void* src, int ld, bool trans, bool f32s,
                                     int m0, int k0, int lane) {
  FragU f;
  const int m = m0 + (lane & 15);
  const int half = lane >> 4;
  if (!trans) {
    if (!f32s) {  // contiguous bf16: 2 x b128 LDS loads
      const unsigned short* s =
          (const unsigned short*)src + (size_t)m * ld + k0 + (half << 3);
      f.q[0] = *(const uint4*)(s);
      f.q[1] = *(const uint4*)(s + 16);
    } else {      // contiguous f32: 4 x b128 loads + pack
      const float* s = (const float*)src + (size_t)m * ld + k0 + (half << 3);
#pragma unroll
      for (int g = 0; g < 2; ++g) {
        const float4 a0 = *(const float4*)(s + g * 16);
        const float4 a1 = *(const float4*)(s + g * 16 + 4);
        f.u[8 * g + 0] = f2bf(a0.x); f.u[8 * g + 1] = f2bf(a0.y);
        f.u[8 * g + 2] = f2bf(a0.z); f.u[8 * g + 3] = f2bf(a0.w);
        f.u[8 * g + 4] = f2bf(a1.x); f.u[8 * g + 5] = f2bf(a1.y);
        f.u[8 * g + 6] = f2bf(a1.z); f.u[8 * g + 7] = f2bf(a1.w);
      }
    }
    return f.v;
  }
  // transposed source: strided gathers (pairs at stride ld -> 2addr loads)
#pragma unroll
  for (int j = 0; j < 8; ++j) {
    const int kk = k0 + ((j >> 2) << 4) + (half << 3) + ((j & 3) << 1);
    const int i0 = kk * ld + m;
    const int i1 = i0 + ld;
    f.u[2 * j]     = f32s ? f2bf(((const float*)src)[i0]) : ((const unsigned short*)src)[i0];
    f.u[2 * j + 1] = f32s ? f2bf(((const float*)src)[i1]) : ((const unsigned short*)src)[i1];
  }
  return f.v;
}

// B (32x16, KxN): lane l -> N=l%16, half=l/16; gather path: vgpr j holds k = 16*half + 2*j
// trans==true means buffer stored [N][K]: per-lane data is two contiguous
// 8-element runs [k0+8h .. +7] and [+16 ..] -> 2 x b128 LDS loads.
__device__ __forceinline__ v16bf ldB(const void* src, int ld, bool trans, bool f32s,
                                     int k0, int n0, int lane) {
  FragU f;
  const int n = n0 + (lane & 15);
  const int half = lane >> 4;
  if (trans && !f32s) {
    const unsigned short* s =
        (const unsigned short*)src + (size_t)n * ld + k0 + (half << 3);
    f.q[0] = *(const uint4*)(s);
    f.q[1] = *(const uint4*)(s + 16);
    return f.v;
  }
#pragma unroll
  for (int j = 0; j < 8; ++j) {
    const int kk = k0 + (half << 4) + (j << 1);
    const int i0 = trans ? (n * ld + kk) : (kk * ld + n);
    const int i1 = trans ? (n * ld + kk + 1) : ((kk + 1) * ld + n);
    f.u[2 * j]     = f32s ? f2bf(((const float*)src)[i0]) : ((const unsigned short*)src)[i0];
    f.u[2 * j + 1] = f32s ? f2bf(((const float*)src)[i1]) : ((const unsigned short*)src)[i1];
  }
  return f.v;
}

// One 16x16 tile of C = A(64x64) * B(64x64); 16 waves, wave w owns tile w.
__device__ __forceinline__ v8f mm64(const void* A, bool ta, bool af,
                                    const void* Bm, bool tb, bool bfp,
                                    int lane, int w) {
  const int m0 = (w >> 2) << 4;
  const int n0 = (w & 3) << 4;
  v8f cc = {0.f, 0.f, 0.f, 0.f, 0.f, 0.f, 0.f, 0.f};
#pragma unroll
  for (int k0 = 0; k0 < 64; k0 += 32) {
    const v16bf av = ldA(A, 64, ta, af, m0, k0, lane);
    const v16bf bv = ldB(Bm, 64, tb, bfp, k0, n0, lane);
    cc = __builtin_amdgcn_wmma_f32_16x16x32_bf16(false, av, false, bv,
                                                 (short)0, cc, false, false);
  }
  return cc;
}

// Scatter this wave's D fragment (f32) to a 64x64 LDS matrix.
__device__ __forceinline__ void st64(float* dst, int lane, int w, const v8f acc) {
  const int mb = ((w >> 2) << 4) + ((lane >> 4) << 3);
  const int n  = ((w & 3) << 4) + (lane & 15);
#pragma unroll
  for (int j = 0; j < 8; ++j) dst[(mb + j) * 64 + n] = acc[j];
}

// Column-wise softmax over the 64 rows (n axis); one thread per column.
// Emits a bf16 copy of p in NORMAL [n][D] layout (scattered, bank-safe).
__device__ __forceinline__ void softmax_col(float* zp, unsigned short* pb, int col) {
  float mx = -3.0e38f;
  for (int n = 0; n < 64; ++n) mx = fmaxf(mx, zp[n * 64 + col]);
  float s = 0.f;
  for (int n = 0; n < 64; ++n) {
    const float e = __expf(zp[n * 64 + col] - mx);
    zp[n * 64 + col] = e;
    s += e;
  }
  const float inv = 1.f / s;
  for (int n = 0; n < 64; ++n) {
    const float p = zp[n * 64 + col] * inv;
    zp[n * 64 + col] = p;
    pb[n * 64 + col] = f2bf(p);
  }
}

// Softmax that also emits p^T bf16 ([D][n] layout) with packed b128 stores.
__device__ __forceinline__ void softmax_col_T(float* zp, unsigned short* pT, int col) {
  float mx = -3.0e38f;
  for (int n = 0; n < 64; ++n) mx = fmaxf(mx, zp[n * 64 + col]);
  float s = 0.f;
  for (int n = 0; n < 64; ++n) {
    const float e = __expf(zp[n * 64 + col] - mx);
    zp[n * 64 + col] = e;
    s += e;
  }
  const float inv = 1.f / s;
  for (int n0 = 0; n0 < 64; n0 += 8) {
    FragU t;
#pragma unroll
    for (int i = 0; i < 8; ++i) {
      const float p = zp[(n0 + i) * 64 + col] * inv;
      zp[(n0 + i) * 64 + col] = p;
      t.u[i] = f2bf(p);
    }
    *(uint4*)&pT[col * 64 + n0] = t.q[0];   // row `col` of p^T, contiguous
  }
}

// ---- f32 -> bf16 bulk convert --------------------------------------------
__global__ void cvt_kernel(const float* __restrict__ in,
                           unsigned short* __restrict__ out, int n4) {
  int i = blockIdx.x * blockDim.x + threadIdx.x;
  const int stride = gridDim.x * blockDim.x;
  for (; i < n4; i += stride) {
    const float4 v = ((const float4*)in)[i];
    uint2 r;
    r.x = (unsigned)f2bf(v.x) | ((unsigned)f2bf(v.y) << 16);
    r.y = (unsigned)f2bf(v.z) | ((unsigned)f2bf(v.w) << 16);
    ((uint2*)out)[i] = r;
  }
}

// ---- bf16 GEMM: C[M,N] = act(A[M,K] @ W[N,K]^T) ---------------------------
// ACT: 0 = none -> f32 out, 1 = silu -> bf16 out, 2 = none -> bf16 out
template <int ACT>
__global__ __launch_bounds__(256) void gemm_bf16(const unsigned short* __restrict__ A,
                                                 const unsigned short* __restrict__ W,
                                                 void* __restrict__ out,
                                                 int M, int N, int K) {
  __shared__ __align__(16) unsigned short sA[128 * 32];
  __shared__ __align__(16) unsigned short sW[128 * 32];
  const int tid  = threadIdx.x;
  const int lane = tid & 31;
  const int w    = tid >> 5;
  const int wm   = w & 3;    // 32-row slab
  const int wn   = w >> 2;   // 64-col slab
  const int m0   = blockIdx.y * 128;
  const int n0   = blockIdx.x * 128;

  v8f acc[2][4];
  const v8f zero = {0.f, 0.f, 0.f, 0.f, 0.f, 0.f, 0.f, 0.f};
#pragma unroll
  for (int mi = 0; mi < 2; ++mi)
#pragma unroll
    for (int ni = 0; ni < 4; ++ni) acc[mi][ni] = zero;

  for (int kt = 0; kt < K; kt += 32) {
    for (int i = tid; i < 512; i += 256) {
      const int row = i >> 2, seg = (i & 3) << 3;
      *(uint4*)&sA[row * 32 + seg] = *(const uint4*)&A[(size_t)(m0 + row) * K + kt + seg];
      *(uint4*)&sW[row * 32 + seg] = *(const uint4*)&W[(size_t)(n0 + row) * K + kt + seg];
    }
    if (kt + 32 < K) {  // hint next K tile into cache (global_prefetch_b8)
      __builtin_prefetch(&A[(size_t)(m0 + (tid >> 1)) * K + kt + 32], 0, 1);
      __builtin_prefetch(&W[(size_t)(n0 + (tid >> 1)) * K + kt + 32], 0, 1);
    }
    __syncthreads();
    v16bf af[2];
#pragma unroll
    for (int mi = 0; mi < 2; ++mi)
      af[mi] = ldA(sA, 32, false, false, wm * 32 + mi * 16, 0, lane);
#pragma unroll
    for (int ni = 0; ni < 4; ++ni) {
      const v16bf bv = ldB(sW, 32, true, false, 0, wn * 64 + ni * 16, lane);
#pragma unroll
      for (int mi = 0; mi < 2; ++mi)
        acc[mi][ni] = __builtin_amdgcn_wmma_f32_16x16x32_bf16(false, af[mi], false, bv,
                                                              (short)0, acc[mi][ni],
                                                              false, false);
    }
    __syncthreads();
  }

  const int mrow = m0 + wm * 32;
  const int ncol = n0 + wn * 64;
#pragma unroll
  for (int mi = 0; mi < 2; ++mi)
#pragma unroll
    for (int ni = 0; ni < 4; ++ni)
#pragma unroll
      for (int j = 0; j < 8; ++j) {
        const int m = mrow + mi * 16 + j + ((lane >> 4) << 3);
        const int n = ncol + ni * 16 + (lane & 15);
        float v = acc[mi][ni][j];
        if (ACT == 1) v = v / (1.f + __expf(-v));     // silu
        if (ACT == 0) ((float*)out)[(size_t)m * N + n] = v;
        else          ((unsigned short*)out)[(size_t)m * N + n] = f2bf(v);
      }
}

// ---- per-chunk lr/wd means (tiny: f32 VALU) -------------------------------
__global__ __launch_bounds__(256) void lrwd_kernel(const float* __restrict__ x,
                                                   const float* __restrict__ Wlr,
                                                   const float* __restrict__ Wb,
                                                   float* __restrict__ lrc,
                                                   float* __restrict__ wdc) {
  __shared__ float red[256];
  const int b = blockIdx.x & (Bn - 1);
  const int c = blockIdx.x / Bn;
  const int tid = threadIdx.x;
  const int j  = tid & 63;   // 0..31 -> lr channel, 32..63 -> wd channel
  const int rg = tid >> 6;   // row group 0..3
  const float* wrow = (j < 32) ? (Wlr + (size_t)j * DIMn)
                               : (Wb  + (size_t)(j - 32) * DIMn);
  float accum = 0.f;
  for (int r = rg; r < CH; r += 4) {
    const float* xr = x + ((size_t)b * Ln + c * CH + r) * DIMn;
    float dot = 0.f;
    for (int kk = 0; kk < DIMn; kk += 4) {
      const float4 xv = *(const float4*)(xr + kk);
      const float4 wv = *(const float4*)(wrow + kk);
      dot = fmaf(xv.x, wv.x, dot);
      dot = fmaf(xv.y, wv.y, dot);
      dot = fmaf(xv.z, wv.z, dot);
      dot = fmaf(xv.w, wv.w, dot);
    }
    accum += 1.f / (1.f + __expf(-dot));
  }
  red[tid] = accum;
  __syncthreads();
  if (tid < 64) {
    float s = red[tid] + red[tid + 64] + red[tid + 128] + red[tid + 192];
    s *= (1.f / 64.f);
    const size_t o = (size_t)(c * Bn + b) * 32;
    if (j < 32) lrc[o + j]        = s * LR0;
    else        wdc[o + (j - 32)] = s * WD0;
  }
}

// ---- sequential chunk scan: 1 block per (b,h), 16 waves, 1 tile per wave --
__global__ __launch_bounds__(512) void scan_kernel(const unsigned short* __restrict__ qb,
                                                   const unsigned short* __restrict__ kb,
                                                   const unsigned short* __restrict__ vb,
                                                   const float* __restrict__ Winit,
                                                   const float* __restrict__ Woinit,
                                                   const float* __restrict__ lrc,
                                                   const float* __restrict__ wdc,
                                                   unsigned short* __restrict__ ob) {
  __shared__ __align__(16) unsigned short sWinB[64 * 64];   // W_in [D][d]; tail: p^T / g_z^T
  __shared__ __align__(16) unsigned short sWoutB[64 * 64];  // bf16 W_out [D][d]
  __shared__ __align__(16) unsigned short sA16[64 * 64];    // q tile, then k tile
  __shared__ __align__(16) unsigned short sV[64 * 64];      // v tile
  __shared__ __align__(16) float sZP[64 * 64];              // logits -> p (f32); tail: [vT|kT] bf16
  __shared__ __align__(16) float sAG[64 * 64];              // q-phase: [p_q bf16 | WoutT bf16]
                                                            // k-phase: a -> g_z (f32)

  const int tid  = threadIdx.x;
  const int lane = tid & 31;
  const int w    = tid >> 5;          // 0..15, one tile per wave
  const int b    = blockIdx.x >> 4;
  const int h    = blockIdx.x & 15;

  const int md = ((w >> 2) << 4) + ((lane >> 4) << 3);  // base row (D), +j
  const int nd = ((w & 3) << 4) + (lane & 15);          // column (d)

  // master fast weights in registers, mapped exactly like WMMA D-fragments
  v8f Wir, Wor;
#pragma unroll
  for (int j = 0; j < 8; ++j) {
    Wir[j] = Winit[((size_t)(md + j) * Hn + h) * Dh + nd];
    Wor[j] = Woinit[((size_t)(md + j) * Hn + h) * Dh + nd];
  }

  unsigned short* pqb     = (unsigned short*)sAG;           // q-phase p_q (bf16)
  unsigned short* sWoutBT = (unsigned short*)sAG + 64 * 64; // q-phase W_out^T (bf16)
  unsigned short* sT      = sWinB;                          // tail: p^T then g_z^T
  unsigned short* sVT     = (unsigned short*)sZP;           // tail: v^T bf16
  unsigned short* sKT     = (unsigned short*)sZP + 64 * 64; // tail: k^T bf16

  for (int c = 0; c < NC; ++c) {
    const size_t sb = (size_t)(c * Bn + b) * 32;
    const float li = lrc[sb + h],      lo = lrc[sb + 16 + h];
    const float wi = wdc[sb + h],      wo = wdc[sb + 16 + h];

    // stage bf16 weights from registers; also W_out^T (one b128 store)
    FragU wtmp;
#pragma unroll
    for (int j = 0; j < 8; ++j) {
      sWinB[(md + j) * 64 + nd]  = f2bf(Wir[j]);
      const unsigned short wob16 = f2bf(Wor[j]);
      sWoutB[(md + j) * 64 + nd] = wob16;
      wtmp.u[j] = wob16;
    }
    *(uint4*)&sWoutBT[nd * 64 + md] = wtmp.q[0];   // [d][D] layout

    // load q + v tiles (64x64 bf16 each); one uint4 per thread per tile
    const size_t rowbase = ((size_t)(b * Ln + c * CH) * Hn + h) * Dh;
    {
      const int n = tid >> 3, seg = (tid & 7) << 3;
      const size_t go = rowbase + (size_t)n * (Hn * Dh) + seg;
      *(uint4*)&sA16[n * 64 + seg] = *(const uint4*)&qb[go];
      *(uint4*)&sV[n * 64 + seg]   = *(const uint4*)&vb[go];
    }
    __syncthreads();

    // z_q[n,D] = sum_d q[n,d] * W_in[D,d]   (A,B both contiguous b128 paths)
    v8f acc = mm64(sA16, false, false, sWinB, true, false, lane, w);
    st64(sZP, lane, w, acc);
    __syncthreads();

    // softmax over n per column D (emit bf16 p_q); overwrite sA16 with k tile
    if (tid < 64) softmax_col(sZP, pqb, tid);
    {
      const int n = tid >> 3, seg = (tid & 7) << 3;
      *(uint4*)&sA16[n * 64 + seg] =
          *(const uint4*)&kb[rowbase + (size_t)n * (Hn * Dh) + seg];
    }
    __syncthreads();

    // o[n,d] = sum_D p_q[n,D] * W_out[D,d]  (A = bf16 p_q, B = W_out^T: b128)
    acc = mm64(pqb, false, false, sWoutBT, true, false, lane, w);
#pragma unroll
    for (int j = 0; j < 8; ++j)
      ob[((size_t)(b * Ln + c * CH + md + j) * Hn + h) * Dh + nd] = f2bf(acc[j]);
    __syncthreads();

    // z_k  (last consumer of sWinB this chunk; buffer then reused as sT)
    acc = mm64(sA16, false, false, sWinB, true, false, lane, w);
    st64(sZP, lane, w, acc);
    __syncthreads();

    // softmax -> p (f32 in sZP) and p^T bf16 into sT (packed b128 rows)
    if (tid < 64) softmax_col_T(sZP, sT, tid);
    __syncthreads();

    // a[n,D] = sum_d W_out[D,d] * v[n,d]
    acc = mm64(sV, false, false, sWoutB, true, false, lane, w);
    st64(sAG, lane, w, acc);
    __syncthreads();

    // g_z[n,D] = p[n,D] * (s[n] - a[n,D]);  s[n] = sum_D p*a   (bank-staggered)
    if (tid < 64) {
      const int row = tid;
      float s = 0.f;
      for (int dd = 0; dd < 64; ++dd) {
        const int D = (dd + row) & 63;
        s += sZP[row * 64 + D] * sAG[row * 64 + D];
      }
      for (int dd = 0; dd < 64; ++dd) {
        const int D = (dd + row) & 63;
        sAG[row * 64 + D] = sZP[row * 64 + D] * (s - sAG[row * 64 + D]);
      }
    }
    __syncthreads();

    // p (f32) is dead: transpose v and k into bf16 [d][n] in sZP's two halves
    if (tid < 128) {
      const int colc = tid & 63;
      const unsigned short* srcm = (tid < 64) ? sV : sA16;
      unsigned short* dstm       = (tid < 64) ? sVT : sKT;
      for (int n0 = 0; n0 < 64; n0 += 8) {
        FragU t;
#pragma unroll
        for (int i = 0; i < 8; ++i) t.u[i] = srcm[(n0 + i) * 64 + colc];
        *(uint4*)&dstm[colc * 64 + n0] = t.q[0];
      }
    }
    __syncthreads();

    // W_out = wo*W_out + lo*(p^T v)   (A = p^T, B = v^T: all b128 paths)
    acc = mm64(sT, false, false, sVT, true, false, lane, w);
    Wor = wo * Wor + lo * acc;
    __syncthreads();

    // transpose g_z -> bf16 g_z^T in sT (column reads bank-safe, b128 stores)
    if (tid < 64) {
      const int Dc = tid;
      for (int n0 = 0; n0 < 64; n0 += 8) {
        FragU t;
#pragma unroll
        for (int i = 0; i < 8; ++i) t.u[i] = f2bf(sAG[(n0 + i) * 64 + Dc]);
        *(uint4*)&sT[Dc * 64 + n0] = t.q[0];
      }
    }
    __syncthreads();

    // W_in = wi*W_in - li*(g_z^T k)   (A = g_z^T, B = k^T: all b128 paths)
    acc = mm64(sT, false, false, sKT, true, false, lane, w);
    Wir = wi * Wir - li * acc;
    __syncthreads();
  }
}

} // namespace

extern "C" void kernel_launch(void* const* d_in, const int* in_sizes, int n_in,
                              void* d_out, int out_size, void* d_ws, size_t ws_size,
                              hipStream_t stream) {
  (void)in_sizes; (void)n_in; (void)out_size; (void)ws_size;
  const float* x      = (const float*)d_in[0];
  const float* Wq     = (const float*)d_in[1];
  const float* Wk     = (const float*)d_in[2];
  const float* Wv     = (const float*)d_in[3];
  const float* Wlr    = (const float*)d_in[4];
  const float* Wbeta  = (const float*)d_in[5];
  const float* Wo     = (const float*)d_in[6];
  const float* Winit  = (const float*)d_in[7];
  const float* Woinit = (const float*)d_in[8];

  char* ws = (char*)d_ws;
  size_t off = 0;
  auto alloc = [&](size_t bytes) {
    void* p = ws + off;
    off += (bytes + 255) & ~(size_t)255;
    return p;
  };
  const size_t MKe = (size_t)Bn * Ln * DIMn;   // 16,777,216 elements
  const size_t We  = (size_t)DIMn * DIMn;      //  1,048,576 elements

  unsigned short* xb  = (unsigned short*)alloc(MKe * 2);
  unsigned short* Wqb = (unsigned short*)alloc(We * 2);
  unsigned short* Wkb = (unsigned short*)alloc(We * 2);
  unsigned short* Wvb = (unsigned short*)alloc(We * 2);
  unsigned short* Wob = (unsigned short*)alloc(We * 2);
  unsigned short* qbb = (unsigned short*)alloc(MKe * 2);
  unsigned short* kbb = (unsigned short*)alloc(MKe * 2);
  unsigned short* vbb = (unsigned short*)alloc(MKe * 2);
  unsigned short* obb = (unsigned short*)alloc(MKe * 2);
  float* lrc = (float*)alloc((size_t)NC * Bn * 32 * sizeof(float));
  float* wdc = (float*)alloc((size_t)NC * Bn * 32 * sizeof(float));

  // 1) f32 -> bf16 conversions (x and GEMM weights)
  cvt_kernel<<<1024, 256, 0, stream>>>(x,  xb,  (int)(MKe / 4));
  cvt_kernel<<<256,  256, 0, stream>>>(Wq, Wqb, (int)(We / 4));
  cvt_kernel<<<256,  256, 0, stream>>>(Wk, Wkb, (int)(We / 4));
  cvt_kernel<<<256,  256, 0, stream>>>(Wv, Wvb, (int)(We / 4));
  cvt_kernel<<<256,  256, 0, stream>>>(Wo, Wob, (int)(We / 4));

  // 2) projections (WMMA bf16)
  const int M = Bn * Ln;                 // 16384
  dim3 ggrid(DIMn / 128, M / 128);       // 8 x 128 blocks
  gemm_bf16<1><<<ggrid, 256, 0, stream>>>(xb, Wqb, qbb, M, DIMn, DIMn);  // silu
  gemm_bf16<1><<<ggrid, 256, 0, stream>>>(xb, Wkb, kbb, M, DIMn, DIMn);  // silu
  gemm_bf16<2><<<ggrid, 256, 0, stream>>>(xb, Wvb, vbb, M, DIMn, DIMn);  // identity

  // 3) chunk-mean lr / wd gates (f32)
  lrwd_kernel<<<NC * Bn, 256, 0, stream>>>(x, Wlr, Wbeta, lrc, wdc);

  // 4) sequential fast-weight scan, one WG per (b,h), 16 waves each
  scan_kernel<<<Bn * Hn, 512, 0, stream>>>(qbb, kbb, vbb, Winit, Woinit,
                                           lrc, wdc, obb);

  // 5) output projection -> f32 d_out
  gemm_bf16<0><<<ggrid, 256, 0, stream>>>(obb, Wob, (float*)d_out, M, DIMn, DIMn);
}
